// MPNNNet_90134183674511
// MI455X (gfx1250) — matrix operand order repaired and checked by reading, not compiled, not run on hardware
//
#include <hip/hip_runtime.h>
#include <hip/hip_bf16.h>

// ---------------------------------------------------------------------------
// MPNN forward for MI455X (gfx1250, wave32, WMMA + TDM).
// Heavy GEMMs (edge-net) run as bf16 WMMA/f32-acc; the per-edge matmul is
// restructured into a per-NODE precompute M = h @ W' (5x less work, M stays
// L2-resident), followed by a VALU gather-contract-scatter over edges.
// B-operands are pre-swizzled into WMMA fragment order so fragment loads are
// wide b128s; W' is staged into LDS via the Tensor Data Mover (6-arg builtin).
// ---------------------------------------------------------------------------

typedef __attribute__((ext_vector_type(16))) __bf16   v16bf;
typedef __attribute__((ext_vector_type(8)))  __bf16   v8bf;
typedef __attribute__((ext_vector_type(8)))  float    v8f;
typedef __attribute__((ext_vector_type(4)))  unsigned uint32x4;
typedef __attribute__((ext_vector_type(8)))  int      int32x8;
typedef __attribute__((ext_vector_type(4)))  int      int32x4;

#if __has_builtin(__builtin_amdgcn_tensor_load_to_lds) && \
    __has_builtin(__builtin_amdgcn_s_wait_tensorcnt)
#define USE_TDM 1
#else
#define USE_TDM 0
#endif

constexpr int V = 20000, E = 100000, D = 32, FN = 36, FE = 12, G = 1000;
constexpr int DEPTH = 3, S2S = 6;
constexpr int NC = 2 * D * D + D;   // 2080: 64*32 weight cols + 32 bias cols of M
constexpr int NT2 = NC / 16;        // 130 B-tiles of M-GEMM
constexpr int W2S_BYTES = D * NC * 2;  // 133,120 B staged in LDS (WGP has 320KB)

__device__ __forceinline__ float    bf2f(__bf16 x) { return (float)x; }
__device__ __forceinline__ __bf16   f2bf(float x)  { return (__bf16)x; }
__device__ __forceinline__ float    sigm(float x)  { return 1.f / (1.f + expf(-x)); }
__device__ __forceinline__ unsigned flipf(float f) {
  unsigned u = __float_as_uint(f);
  return (u & 0x80000000u) ? ~u : (u | 0x80000000u);
}
__device__ __forceinline__ float unflipf(unsigned u) {
  return __uint_as_float((u & 0x80000000u) ? (u ^ 0x80000000u) : ~u);
}

// --- WMMA fragment loaders (16x16x32 bf16, contiguous k order used for both A and B) -------
// A (16x32, MxK): lane L -> row M=L&15, hi=L>>4. Elements are the contiguous
// run K=[16*hi, 16*hi+16): two 16-byte loads.
__device__ __forceinline__ v16bf load_frag_A(const __bf16* A, int lda) {
  int lane = threadIdx.x & 31;
  int m = lane & 15, hi = lane >> 4;
  const __bf16* row = A + m * lda + 16 * hi;
  v8bf a0 = *(const v8bf*)(row);
  v8bf a1 = *(const v8bf*)(row + 8);
  return __builtin_shufflevector(a0, a1, 0, 1, 2, 3, 4, 5, 6, 7,
                                 8, 9, 10, 11, 12, 13, 14, 15);
}
// B tiles pre-swizzled to fragment order: tile[lane][e] (32B/lane contiguous).
__device__ __forceinline__ v16bf load_frag_Bswz(const __bf16* tile) {
  const __bf16* p = tile + (threadIdx.x & 31) * 16;
  v8bf b0 = *(const v8bf*)(p);
  v8bf b1 = *(const v8bf*)(p + 8);
  return __builtin_shufflevector(b0, b1, 0, 1, 2, 3, 4, 5, 6, 7,
                                 8, 9, 10, 11, 12, 13, 14, 15);
}

// --- Embedding -------------------------------------------------------------
__global__ void k_embed_nodes(const float* __restrict__ node, const float* __restrict__ Wn,
                              const float* __restrict__ bn, float* __restrict__ h,
                              float* __restrict__ h0) {
  int v = (blockIdx.x * blockDim.x + threadIdx.x) >> 5;
  if (v >= V) return;
  int j = threadIdx.x & 31;
  float s = bn[j];
  const float* row = node + (size_t)v * FN;
#pragma unroll
  for (int t = 0; t < FN; ++t) s += row[t] * Wn[t * D + j];
  s = s > 0.f ? s : 0.01f * s;
  h[v * D + j] = s;
  h0[v * D + j] = s;
}

__global__ void k_embed_edges(const float* __restrict__ edge, const float* __restrict__ We,
                              const float* __restrict__ be, __bf16* __restrict__ ehbf) {
  int e = (blockIdx.x * blockDim.x + threadIdx.x) >> 5;
  if (e >= E) return;
  int j = threadIdx.x & 31;
  float s = be[j];
  const float* row = edge + (size_t)e * FE;
#pragma unroll
  for (int t = 0; t < FE; ++t) s += row[t] * We[t * D + j];
  s = s > 0.f ? s : 0.01f * s;
  ehbf[e * D + j] = f2bf(s);
}

// --- Per-layer weight prep: convert to bf16 in WMMA-B fragment order -------
// Fragment order: index = ((nt*32 + lane)*16 + e); element = W[K=16*hi+e, nt*16+n]
__global__ void k_prep_w1(const float* __restrict__ eW1l, __bf16* __restrict__ w1s) {
  int i = blockIdx.x * blockDim.x + threadIdx.x;
  if (i >= D * 2 * D) return;
  int e = i & 15, lane = (i >> 4) & 31, nt = i >> 9;
  int n = lane & 15, hi = lane >> 4;
  int K = 16 * hi + e, col = nt * 16 + n;
  w1s[i] = f2bf(eW1l[K * (2 * D) + col]);
}
// W'[d, k*32+f] = eW2[k, d*32+f]; W'[d, 2048+f] = eb2[d*32+f]
__global__ void k_prep_w2(const float* __restrict__ eW2l, const float* __restrict__ eb2l,
                          __bf16* __restrict__ w2s) {
  int i = blockIdx.x * blockDim.x + threadIdx.x;
  if (i >= D * NC) return;
  int e = i & 15, lane = (i >> 4) & 31, nt = i >> 9;
  int n = lane & 15, hi = lane >> 4;
  int K = 16 * hi + e, col = nt * 16 + n;
  float v;
  if (col < 2 * D * D) { int k = col / D, f = col % D; v = eW2l[k * (D * D) + K * D + f]; }
  else                 { int f = col - 2 * D * D;      v = eb2l[K * D + f]; }
  w2s[i] = f2bf(v);
}
__global__ void k_h2bf(const float* __restrict__ h, __bf16* __restrict__ hbf) {
  int i = blockIdx.x * blockDim.x + threadIdx.x;
  if (i < V * D) hbf[i] = f2bf(h[i]);
}

// --- a = relu(e_h @ eW1 + eb1) : (E x 32)@(32 x 64), WMMA ------------------
__global__ void k_edge_a(const __bf16* __restrict__ ehbf, const __bf16* __restrict__ w1s,
                         const float* __restrict__ eb1l, float* __restrict__ a) {
  int tile = blockIdx.x * (blockDim.x >> 5) + (threadIdx.x >> 5);
  if (tile >= E / 16) return;
  int lane = threadIdx.x & 31;
  int n = lane & 15, hi = lane >> 4;
  v16bf af = load_frag_A(ehbf + (size_t)tile * 16 * D, D);
#pragma unroll
  for (int nt = 0; nt < 4; ++nt) {
    v16bf bf = load_frag_Bswz(w1s + nt * 512);
    v8f acc = {};
    acc = __builtin_amdgcn_wmma_f32_16x16x32_bf16(false, af, false, bf, (short)0, acc,
                                                  false, false);
    int col = nt * 16 + n;
    float bias = eb1l[col];
#pragma unroll
    for (int r = 0; r < 8; ++r) {
      float vv = acc[r] + bias;
      a[(size_t)(tile * 16 + hi * 8 + r) * (2 * D) + col] = vv > 0.f ? vv : 0.f;
    }
  }
}

// --- M = hbf @ W' : (V x 32)@(32 x 2080), WMMA; W' TDM-staged in LDS -------
__global__ void k_node_M(const __bf16* __restrict__ hbf, const __bf16* __restrict__ w2s,
                         __bf16* __restrict__ M2) {
  extern __shared__ __bf16 sW[];  // W2S_BYTES = 133,120 bytes
#if USE_TDM
  if ((threadIdx.x >> 5) == 0) {
    // 1-D D# descriptor: copy D*NC bf16 (16640 x 8B) global -> LDS via TDM.
    unsigned nElem = (unsigned)(W2S_BYTES / 8);  // 16640 8-byte elements
    unsigned lds = (unsigned)(size_t)(void*)sW;
    unsigned long long ga = (unsigned long long)(size_t)(const void*)w2s;
    uint32x4 g0;
    g0[0] = 1u;                                   // count=1 (valid user D#)
    g0[1] = lds;                                  // lds_addr
    g0[2] = (unsigned)(ga & 0xffffffffu);         // global_addr[31:0]
    g0[3] = (unsigned)((ga >> 32) & 0x01ffffffu)  // global_addr[56:32]
            | (2u << 30);                         // type=2 ("image")
    int32x8 g1;
    g1[0] = (int)(3u << 16);                      // data_size=3 (8B), no extras
    g1[1] = (int)((nElem & 0xffffu) << 16);       // tensor_dim0[15:0] @ bit48
    g1[2] = (int)((nElem >> 16) & 0xffffu);       // tensor_dim0[31:16]
    g1[3] = (int)((nElem & 0xffffu) << 16);       // tile_dim0 @ bit112
    g1[4] = 0;                                    // tile_dim1/2 unused (1-D)
    g1[5] = (int)nElem;                           // tensor_dim0_stride[31:0]
    g1[6] = 0;
    g1[7] = 0;
    int32x4 z4 = {0, 0, 0, 0};
    int32x8 z8 = {0, 0, 0, 0, 0, 0, 0, 0};
    __builtin_amdgcn_tensor_load_to_lds(g0, g1, z4, z4, z8, 0);
    __builtin_amdgcn_s_wait_tensorcnt(0);
  }
  __syncthreads();
#else
  {
    const unsigned* s = (const unsigned*)w2s;
    unsigned* d = (unsigned*)sW;
    for (int i = threadIdx.x; i < (D * NC) / 2; i += blockDim.x) d[i] = s[i];
  }
  __syncthreads();
#endif
  int tile = blockIdx.x * (blockDim.x >> 5) + (threadIdx.x >> 5);
  if (tile >= V / 16) return;
  int lane = threadIdx.x & 31;
  int n = lane & 15, hi = lane >> 4;
  v16bf af = load_frag_A(hbf + (size_t)tile * 16 * D, D);
  for (int nt = 0; nt < NT2; ++nt) {
    v16bf bf = load_frag_Bswz(sW + nt * 512);
    v8f acc = {};
    acc = __builtin_amdgcn_wmma_f32_16x16x32_bf16(false, af, false, bf, (short)0, acc,
                                                  false, false);
#pragma unroll
    for (int r = 0; r < 8; ++r)
      M2[(size_t)(tile * 16 + hi * 8 + r) * NC + nt * 16 + n] = f2bf(acc[r]);
  }
}

// --- msg[e] = a[e] (1x64) . M[src[e]] (64x32) + bias; scatter-add to dst ---
constexpr int EPW = 4;  // edges per wave (lets us prefetch the next gather)
__global__ void k_message(const float* __restrict__ a, const __bf16* __restrict__ M2,
                          const int* __restrict__ src, const int* __restrict__ dst,
                          float* __restrict__ agg) {
  int w = blockIdx.x * (blockDim.x >> 5) + (threadIdx.x >> 5);
  int e0 = w * EPW;
  if (e0 >= E) return;
  int lane = threadIdx.x & 31;
  for (int e = e0; e < e0 + EPW; ++e) {
    if (e + 1 < e0 + EPW)  // warm L2/L1 for the next edge's M-row gather
      __builtin_prefetch(M2 + (size_t)src[e + 1] * NC + lane * 64, 0, 1);
    float alo = a[(size_t)e * 64 + lane];
    float ahi = a[(size_t)e * 64 + 32 + lane];
    const __bf16* base = M2 + (size_t)src[e] * NC;
    float acc = bf2f(base[2 * D * D + lane]);  // h[src] . eb2 term
#pragma unroll 8
    for (int k = 0; k < 32; ++k) acc += __shfl(alo, k) * bf2f(base[k * D + lane]);
#pragma unroll 8
    for (int k = 0; k < 32; ++k) acc += __shfl(ahi, k) * bf2f(base[(32 + k) * D + lane]);
    atomicAdd(&agg[(size_t)dst[e] * D + lane], acc);
  }
}

// --- GRU(relu(agg), hidden0) + LayerNorm, one wave per node ----------------
__global__ void k_gru(const float* __restrict__ agg, const float* __restrict__ h0,
                      const float* __restrict__ wih, const float* __restrict__ whh,
                      const float* __restrict__ bih, const float* __restrict__ bhh,
                      const float* __restrict__ lng, const float* __restrict__ lnb,
                      float* __restrict__ h) {
  __shared__ float sWih[32 * 96], sWhh[32 * 96];  // transposed [d][c]: conflict-free
  __shared__ float sBih[96], sBhh[96], sG[32], sB[32];
  for (int i = threadIdx.x; i < 96 * 32; i += blockDim.x) {
    int c = i / 32, d = i % 32;
    sWih[d * 96 + c] = wih[i];
    sWhh[d * 96 + c] = whh[i];
  }
  for (int i = threadIdx.x; i < 96; i += blockDim.x) { sBih[i] = bih[i]; sBhh[i] = bhh[i]; }
  for (int i = threadIdx.x; i < 32; i += blockDim.x) { sG[i] = lng[i]; sB[i] = lnb[i]; }
  __syncthreads();
  int v = blockIdx.x * (blockDim.x >> 5) + (threadIdx.x >> 5);
  if (v >= V) return;
  int j = threadIdx.x & 31;
  float aj = agg[(size_t)v * D + j]; aj = aj > 0.f ? aj : 0.f;  // relu on aggregate
  float hj = h0[(size_t)v * D + j];
  float gi0 = sBih[j], gi1 = sBih[32 + j], gi2 = sBih[64 + j];
  float gh0 = sBhh[j], gh1 = sBhh[32 + j], gh2 = sBhh[64 + j];
  for (int d = 0; d < 32; ++d) {
    float ad = __shfl(aj, d), hd = __shfl(hj, d);
    const float* wi = &sWih[d * 96];
    const float* wh = &sWhh[d * 96];
    gi0 += ad * wi[j]; gi1 += ad * wi[32 + j]; gi2 += ad * wi[64 + j];
    gh0 += hd * wh[j]; gh1 += hd * wh[32 + j]; gh2 += hd * wh[64 + j];
  }
  float r = sigm(gi0 + gh0);
  float z = sigm(gi1 + gh1);
  float n = tanhf(gi2 + r * gh2);
  float hn = (1.f - z) * n + z * hj;
  float m = hn;
  for (int off = 16; off; off >>= 1) m += __shfl_xor(m, off);
  m *= (1.f / 32.f);
  float dlt = hn - m, var = dlt * dlt;
  for (int off = 16; off; off >>= 1) var += __shfl_xor(var, off);
  var *= (1.f / 32.f);
  h[(size_t)v * D + j] = dlt * rsqrtf(var + 1e-5f) * sG[j] + sB[j];
}

// --- Set2Set ---------------------------------------------------------------
__global__ void k_lstm(const float* __restrict__ qstar, const float* __restrict__ wih,
                       const float* __restrict__ whh, const float* __restrict__ bih,
                       const float* __restrict__ bhh, float* __restrict__ hh,
                       float* __restrict__ cc) {
  int g = blockIdx.x * (blockDim.x >> 5) + (threadIdx.x >> 5);
  if (g >= G) return;
  int j = threadIdx.x & 31;
  float qlo = qstar[g * 64 + j], qhi = qstar[g * 64 + 32 + j], hj = hh[g * D + j];
  float g0 = bih[j] + bhh[j], g1 = bih[32 + j] + bhh[32 + j];
  float g2 = bih[64 + j] + bhh[64 + j], g3 = bih[96 + j] + bhh[96 + j];
  for (int t = 0; t < 32; ++t) {
    float q = __shfl(qlo, t);
    g0 += q * wih[j * 64 + t];        g1 += q * wih[(32 + j) * 64 + t];
    g2 += q * wih[(64 + j) * 64 + t]; g3 += q * wih[(96 + j) * 64 + t];
  }
  for (int t = 0; t < 32; ++t) {
    float q = __shfl(qhi, t);
    g0 += q * wih[j * 64 + 32 + t];        g1 += q * wih[(32 + j) * 64 + 32 + t];
    g2 += q * wih[(64 + j) * 64 + 32 + t]; g3 += q * wih[(96 + j) * 64 + 32 + t];
  }
  for (int t = 0; t < 32; ++t) {
    float q = __shfl(hj, t);
    g0 += q * whh[j * 32 + t];        g1 += q * whh[(32 + j) * 32 + t];
    g2 += q * whh[(64 + j) * 32 + t]; g3 += q * whh[(96 + j) * 32 + t];
  }
  float ii = sigm(g0), ff = sigm(g1), gg = tanhf(g2), oo = sigm(g3);
  float c = ff * cc[g * D + j] + ii * gg;
  cc[g * D + j] = c;
  hh[g * D + j] = oo * tanhf(c);
}

__global__ void k_att1(const float* __restrict__ h, const float* __restrict__ hh,
                       const int* __restrict__ n2g, float* __restrict__ att,
                       unsigned* __restrict__ mu) {
  int v = blockIdx.x * (blockDim.x >> 5) + (threadIdx.x >> 5);
  if (v >= V) return;
  int j = threadIdx.x & 31;
  int g = n2g[v];
  float s = h[(size_t)v * D + j] * hh[g * D + j];
  for (int off = 16; off; off >>= 1) s += __shfl_xor(s, off);
  if (j == 0) { att[v] = s; atomicMax(&mu[g], flipf(s)); }
}

__global__ void k_att2(const float* __restrict__ att, const int* __restrict__ n2g,
                       const unsigned* __restrict__ mu, float* __restrict__ denom,
                       float* __restrict__ ex) {
  int v = blockIdx.x * blockDim.x + threadIdx.x;
  if (v >= V) return;
  int g = n2g[v];
  float e = expf(att[v] - unflipf(mu[g]));
  ex[v] = e;
  atomicAdd(&denom[g], e);
}

__global__ void k_att3(const float* __restrict__ h, const int* __restrict__ n2g,
                       const float* __restrict__ ex, const float* __restrict__ denom,
                       float* __restrict__ rread) {
  int v = blockIdx.x * (blockDim.x >> 5) + (threadIdx.x >> 5);
  if (v >= V) return;
  int j = threadIdx.x & 31;
  int g = n2g[v];
  float alpha = ex[v] / denom[g];
  atomicAdd(&rread[g * D + j], alpha * h[(size_t)v * D + j]);
}

__global__ void k_qcat(const float* __restrict__ hh, const float* __restrict__ rread,
                       float* __restrict__ qstar) {
  int i = blockIdx.x * blockDim.x + threadIdx.x;
  if (i >= G * 2 * D) return;
  int g = i / (2 * D), t = i % (2 * D);
  qstar[i] = (t < D) ? hh[g * D + t] : rread[g * D + t - D];
}

__global__ void k_final(const float* __restrict__ qstar, const float* __restrict__ Wp1,
                        const float* __restrict__ bp1, const float* __restrict__ Wp2,
                        const float* __restrict__ bp2, float* __restrict__ out) {
  int g = blockIdx.x * (blockDim.x >> 5) + (threadIdx.x >> 5);
  if (g >= G) return;
  int j = threadIdx.x & 31;
  float qlo = qstar[g * 64 + j], qhi = qstar[g * 64 + 32 + j];
  float t = bp1[j];
  for (int u = 0; u < 32; ++u) t += __shfl(qlo, u) * Wp1[u * 32 + j];
  for (int u = 0; u < 32; ++u) t += __shfl(qhi, u) * Wp1[(32 + u) * 32 + j];
  t = t > 0.f ? t : 0.f;
  float s = t * Wp2[j];
  for (int off = 16; off; off >>= 1) s += __shfl_xor(s, off);
  if (j == 0) out[g] = s + bp2[0];
}

// ---------------------------------------------------------------------------
extern "C" void kernel_launch(void* const* d_in, const int* in_sizes, int n_in,
                              void* d_out, int out_size, void* d_ws, size_t ws_size,
                              hipStream_t stream) {
  (void)in_sizes; (void)n_in; (void)out_size; (void)ws_size;
  const float* node = (const float*)d_in[0];
  const float* edge = (const float*)d_in[1];
  const int* srcI   = (const int*)d_in[2];
  const int* dstI   = (const int*)d_in[3];
  const int* n2g    = (const int*)d_in[4];
  const float* Wn = (const float*)d_in[5];   const float* bn = (const float*)d_in[6];
  const float* We = (const float*)d_in[7];   const float* be = (const float*)d_in[8];
  const float* eW1 = (const float*)d_in[9];  const float* eb1 = (const float*)d_in[10];
  const float* eW2 = (const float*)d_in[11]; const float* eb2 = (const float*)d_in[12];
  const float* gwih = (const float*)d_in[13]; const float* gwhh = (const float*)d_in[14];
  const float* gbih = (const float*)d_in[15]; const float* gbhh = (const float*)d_in[16];
  const float* lng = (const float*)d_in[17];  const float* lnb = (const float*)d_in[18];
  const float* lwih = (const float*)d_in[19]; const float* lwhh = (const float*)d_in[20];
  const float* lbih = (const float*)d_in[21]; const float* lbhh = (const float*)d_in[22];
  const float* Wp1 = (const float*)d_in[23];  const float* bp1 = (const float*)d_in[24];
  const float* Wp2 = (const float*)d_in[25];  const float* bp2 = (const float*)d_in[26];
  float* out = (float*)d_out;

  char* p = (char*)d_ws;
  auto carve = [&](size_t bytes) -> char* {
    char* r = p; p += (bytes + 255) & ~(size_t)255; return r;
  };
  float*  h     = (float*)carve((size_t)V * D * 4);
  float*  h0    = (float*)carve((size_t)V * D * 4);
  __bf16* hbf   = (__bf16*)carve((size_t)V * D * 2);
  __bf16* ehbf  = (__bf16*)carve((size_t)E * D * 2);
  __bf16* w1s   = (__bf16*)carve((size_t)D * 2 * D * 2);
  __bf16* w2s   = (__bf16*)carve((size_t)W2S_BYTES);
  float*  aE    = (float*)carve((size_t)E * 2 * D * 4);
  __bf16* M2    = (__bf16*)carve((size_t)V * NC * 2);
  float*  agg   = (float*)carve((size_t)V * D * 4);
  float*  qstar = (float*)carve((size_t)G * 2 * D * 4);
  float*  hhb   = (float*)carve((size_t)G * D * 4);
  float*  ccb   = (float*)carve((size_t)G * D * 4);
  float*  att   = (float*)carve((size_t)V * 4);
  float*  exb   = (float*)carve((size_t)V * 4);
  unsigned* mu  = (unsigned*)carve((size_t)G * 4);
  float*  denom = (float*)carve((size_t)G * 4);
  float*  rread = (float*)carve((size_t)G * D * 4);

  k_embed_nodes<<<V / 8, 256, 0, stream>>>(node, Wn, bn, h, h0);
  k_embed_edges<<<E / 8, 256, 0, stream>>>(edge, We, be, ehbf);

  for (int l = 0; l < DEPTH; ++l) {
    k_prep_w1<<<(D * 2 * D + 255) / 256, 256, 0, stream>>>(eW1 + l * D * 2 * D, w1s);
    k_prep_w2<<<(D * NC + 255) / 256, 256, 0, stream>>>(eW2 + (size_t)l * 2 * D * D * D,
                                                        eb2 + l * D * D, w2s);
    k_h2bf<<<(V * D + 255) / 256, 256, 0, stream>>>(h, hbf);
    k_edge_a<<<(E / 16 + 7) / 8, 256, 0, stream>>>(ehbf, w1s, eb1 + l * 2 * D, aE);
    k_node_M<<<(V / 16 + 7) / 8, 256, W2S_BYTES, stream>>>(hbf, w2s, M2);
    (void)hipMemsetAsync(agg, 0, (size_t)V * D * 4, stream);
    k_message<<<E / EPW / 8, 256, 0, stream>>>(aE, M2, srcI, dstI, agg);
    k_gru<<<V / 8, 256, 0, stream>>>(agg, h0, gwih + l * 96 * D, gwhh + l * 96 * D,
                                     gbih + l * 96, gbhh + l * 96, lng + l * D,
                                     lnb + l * D, h);
  }

  (void)hipMemsetAsync(qstar, 0, (size_t)G * 2 * D * 4, stream);
  (void)hipMemsetAsync(hhb, 0, (size_t)G * D * 4, stream);
  (void)hipMemsetAsync(ccb, 0, (size_t)G * D * 4, stream);
  for (int it = 0; it < S2S; ++it) {
    (void)hipMemsetAsync(mu, 0, (size_t)G * 4, stream);
    (void)hipMemsetAsync(denom, 0, (size_t)G * 4, stream);
    (void)hipMemsetAsync(rread, 0, (size_t)G * D * 4, stream);
    k_lstm<<<G / 8, 256, 0, stream>>>(qstar, lwih, lwhh, lbih, lbhh, hhb, ccb);
    k_att1<<<V / 8, 256, 0, stream>>>(h, hhb, n2g, att, mu);
    k_att2<<<(V + 255) / 256, 256, 0, stream>>>(att, n2g, mu, denom, exb);
    k_att3<<<V / 8, 256, 0, stream>>>(h, n2g, exb, denom, rread);
    k_qcat<<<(G * 2 * D + 255) / 256, 256, 0, stream>>>(hhb, rread, qstar);
  }
  k_final<<<G / 8, 256, 0, stream>>>(qstar, Wp1, bp1, Wp2, bp2, out);
}
